// EquivNet_40003325395261
// MI455X (gfx1250) — matrix operand order repaired and checked by reading, not compiled, not run on hardware
//
#include <hip/hip_runtime.h>
#include <hip/hip_bf16.h>

// ---------------------------------------------------------------------------
// EquivNet (PaiNN-style) forward for MI455X / gfx1250, wave32 + WMMA.
//  - edge MLP chain fused into one WMMA kernel (2 x 16-edge tiles / wave,
//    software-pipelined shared B fragments, preloaded A fragments)
//  - phi MLP and UV-update chain are WMMA kernels too (no scratch spills)
//  - all weights pre-packed to f16 B-fragment order (2 x b128 per fragment)
//  - f16 mirrors of s / v / phi: every gather is a wide f16 load
// ---------------------------------------------------------------------------

typedef __attribute__((ext_vector_type(16))) _Float16 v16h;
typedef __attribute__((ext_vector_type(8)))  _Float16 v8h;
typedef __attribute__((ext_vector_type(4)))  _Float16 v4h;
typedef __attribute__((ext_vector_type(8)))  float    v8f;

#define NL  3
#define NN  40000
#define NE  640000
#define DD  64
#define NG  16

__device__ __forceinline__ float silu_f(float x) { return x / (1.0f + __expf(-x)); }
__device__ __forceinline__ float sigm_f(float x) { return 1.0f / (1.0f + __expf(-x)); }

__device__ __forceinline__ v8h cvt8(v8f x) {
  v8h r;
#pragma unroll
  for (int i = 0; i < 8; ++i) r[i] = (_Float16)x[i];
  return r;
}
__device__ __forceinline__ v16h cat2(v8h lo, v8h hi2) {
  return __builtin_shufflevector(lo, hi2, 0, 1, 2, 3, 4, 5, 6, 7,
                                 8, 9, 10, 11, 12, 13, 14, 15);
}
__device__ __forceinline__ v16h loadB(const _Float16* __restrict__ p, int f, int lane) {
  return *(const v16h*)(p + ((size_t)f << 9) + (lane << 4));
}
__device__ __forceinline__ v16h a_from_f32(const float* __restrict__ row, int kc, int hi) {
  v8f lo = *(const v8f*)(row + kc * 32 + hi * 8);
  v8f hh = *(const v8f*)(row + kc * 32 + 16 + hi * 8);
  return cat2(cvt8(lo), cvt8(hh));
}
__device__ __forceinline__ v16h a_from_lds(const _Float16* __restrict__ row, int kc, int hi) {
  v8h lo = *(const v8h*)(row + kc * 32 + hi * 8);
  v8h hh = *(const v8h*)(row + kc * 32 + 16 + hi * 8);
  return cat2(lo, hh);
}
#define WMMA(A, B, C) __builtin_amdgcn_wmma_f32_16x16x32_f16(false, A, false, B, (short)0, C, false, false)

// 8 consecutive f16 cat-row values starting at K = kbase + hi*8 (kbase is a
// compile-time multiple of 16 after unrolling -> region resolves statically).
__device__ __forceinline__ v8h cat_run(const _Float16* __restrict__ s_h,
                                       int sE, int dE, float de,
                                       const float* __restrict__ efW,
                                       const float* __restrict__ efb,
                                       int kbase, int hi) {
  if (kbase < 64)
    return *(const v8h*)(s_h + (size_t)sE * DD + kbase + hi * 8);
  if (kbase < 128)
    return *(const v8h*)(s_h + (size_t)dE * DD + (kbase - 64) + hi * 8);
  int q0 = kbase - 128 + hi * 8;
  v8f wv = *(const v8f*)(efW + q0);
  v8f bv = *(const v8f*)(efb + q0);
  v8f x;
#pragma unroll
  for (int i = 0; i < 8; ++i) x[i] = silu_f(de * wv[i] + bv[i]);
  return cvt8(x);
}

// ---------------------------------------------------------------------------
// Pack f32 row-major W (K x Ncol) into f16 B-fragment order:
//   frag f = kc*NT + nt ; element (lane, j) at P[f*512 + lane*16 + j]
//   = W[(kc*32 + (j>>3)*16 + (lane>>4)*8 + (j&7)) * Ncol + nt*16 + (lane&15)]
// ---------------------------------------------------------------------------
__global__ void pack_w_kernel(const float* __restrict__ W, _Float16* __restrict__ P,
                              int NT, int Ncol, int total) {
  int idx = blockIdx.x * blockDim.x + threadIdx.x;
  if (idx >= total) return;
  int f   = idx >> 9;
  int rem = idx & 511;
  int l   = rem >> 4, j = rem & 15;
  int kc  = f / NT, nt = f % NT;
  int K   = kc * 32 + (j >> 3) * 16 + (l >> 4) * 8 + (j & 7);
  int col = nt * 16 + (l & 15);
  P[idx] = (_Float16)W[(size_t)K * Ncol + col];
}

// ---------------------------------------------------------------------------
// Fused per-edge pipeline; one wave = two 16-edge tiles, 4 waves / block,
// E % 128 == 0 so EXEC is all-ones everywhere (WMMA requirement).
// ---------------------------------------------------------------------------
__global__ __launch_bounds__(128) void edge_kernel(
    const _Float16* __restrict__ s_h,  const _Float16* __restrict__ v_h,
    const _Float16* __restrict__ phi_h,
    const float* __restrict__ dist, const float* __restrict__ unit,
    const int*   __restrict__ edges,
    const float* __restrict__ efW,  const float* __restrict__ efb,
    const _Float16* __restrict__ pW1, const float* __restrict__ elb1,
    const _Float16* __restrict__ pW2, const float* __restrict__ elb2,
    const _Float16* __restrict__ pW3, const float* __restrict__ inb,
    const float* __restrict__ eiW,  const float* __restrict__ eib,
    float* __restrict__ s_out, float* __restrict__ v_out)
{
  __shared__ _Float16 hbuf[4][2][16][128];   // 32 KB silu(hidden) tiles
  __shared__ _Float16 es2buf[4][2][16][64];  // 16 KB es2 tiles

  const int w    = threadIdx.x >> 5;
  const int lane = threadIdx.x & 31;
  const int m    = lane & 15;
  const int hi   = lane >> 4;
  const int e0   = (blockIdx.x * 4 + w) * 32;

  if (threadIdx.x == 0) {   // gfx1250 global_prefetch_b8 for hot weights
    __builtin_prefetch(pW1, 0, 0);
    __builtin_prefetch(pW2, 0, 0);
    __builtin_prefetch(pW3, 0, 0);
  }

  int   sE[2], dE[2];
  float de[2];
#pragma unroll
  for (int t = 0; t < 2; ++t) {
    int eT = e0 + t * 16 + m;
    sE[t] = edges[2 * eT];
    dE[t] = edges[2 * eT + 1];
    de[t] = dist[eT];
  }
  const v8f zf = {0.f, 0.f, 0.f, 0.f, 0.f, 0.f, 0.f, 0.f};

  // ---- GEMM1: cat(16x192) @ el_W1(192x128), bias folded into silu store ---
  v8f acc1[2][8];
#pragma unroll
  for (int nt = 0; nt < 8; ++nt) { acc1[0][nt] = zf; acc1[1][nt] = zf; }
  v16h a1[6][2];
#pragma unroll
  for (int kc = 0; kc < 6; ++kc)
#pragma unroll
    for (int t = 0; t < 2; ++t)
      a1[kc][t] = cat2(cat_run(s_h, sE[t], dE[t], de[t], efW, efb, kc * 32, hi),
                       cat_run(s_h, sE[t], dE[t], de[t], efW, efb, kc * 32 + 16, hi));
  {
    v16h bcur = loadB(pW1, 0, lane);
#pragma unroll
    for (int kc = 0; kc < 6; ++kc)
#pragma unroll
      for (int nt = 0; nt < 8; ++nt) {
        int f = kc * 8 + nt;
        v16h bnxt = loadB(pW1, (f + 1 < 48) ? f + 1 : f, lane);
        acc1[0][nt] = WMMA(a1[kc][0], bcur, acc1[0][nt]);
        acc1[1][nt] = WMMA(a1[kc][1], bcur, acc1[1][nt]);
        bcur = bnxt;
      }
  }
#pragma unroll
  for (int nt = 0; nt < 8; ++nt) {
    float b0 = elb1[nt * 16 + m];
#pragma unroll
    for (int t = 0; t < 2; ++t)
#pragma unroll
      for (int r = 0; r < 8; ++r)
        hbuf[w][t][hi * 8 + r][nt * 16 + m] = (_Float16)silu_f(acc1[t][nt][r] + b0);
  }
  __syncthreads();

  // ---- GEMM2: h(16x128) @ el_W2(128x64), bias folded into store -----------
  v8f acc2[2][4];
#pragma unroll
  for (int nt = 0; nt < 4; ++nt) { acc2[0][nt] = zf; acc2[1][nt] = zf; }
  v16h a2[4][2];
#pragma unroll
  for (int kc = 0; kc < 4; ++kc)
#pragma unroll
    for (int t = 0; t < 2; ++t)
      a2[kc][t] = a_from_lds(&hbuf[w][t][m][0], kc, hi);
  {
    v16h bcur = loadB(pW2, 0, lane);
#pragma unroll
    for (int kc = 0; kc < 4; ++kc)
#pragma unroll
      for (int nt = 0; nt < 4; ++nt) {
        int f = kc * 4 + nt;
        v16h bnxt = loadB(pW2, (f + 1 < 16) ? f + 1 : f, lane);
        acc2[0][nt] = WMMA(a2[kc][0], bcur, acc2[0][nt]);
        acc2[1][nt] = WMMA(a2[kc][1], bcur, acc2[1][nt]);
        bcur = bnxt;
      }
  }
#pragma unroll
  for (int nt = 0; nt < 4; ++nt) {
    float b0 = elb2[nt * 16 + m];
#pragma unroll
    for (int t = 0; t < 2; ++t)
#pragma unroll
      for (int r = 0; r < 8; ++r)
        es2buf[w][t][hi * 8 + r][nt * 16 + m] = (_Float16)(acc2[t][nt][r] + b0);
  }
  __syncthreads();

  // ---- GEMM3: es2(16x64) @ in_W(64x192), bias folded into epilogue --------
  v8f acc3[2][12];
#pragma unroll
  for (int nt = 0; nt < 12; ++nt) { acc3[0][nt] = zf; acc3[1][nt] = zf; }
  v16h a3[2][2];
#pragma unroll
  for (int kc = 0; kc < 2; ++kc)
#pragma unroll
    for (int t = 0; t < 2; ++t)
      a3[kc][t] = a_from_lds(&es2buf[w][t][m][0], kc, hi);
  {
    v16h bcur = loadB(pW3, 0, lane);
#pragma unroll
    for (int kc = 0; kc < 2; ++kc)
#pragma unroll
      for (int nt = 0; nt < 12; ++nt) {
        int f = kc * 12 + nt;
        v16h bnxt = loadB(pW3, (f + 1 < 24) ? f + 1 : f, lane);
        acc3[0][nt] = WMMA(a3[kc][0], bcur, acc3[0][nt]);
        acc3[1][nt] = WMMA(a3[kc][1], bcur, acc3[1][nt]);
        bcur = bnxt;
      }
  }

  // ---- Wphi = (Wv + b) * phi[src]; gate; gated scatter --------------------
  const float eib0 = eib[0];
  float eiv[4], bb[12];
#pragma unroll
  for (int nt = 0; nt < 4; ++nt) eiv[nt] = eiW[nt * 16 + m];
#pragma unroll
  for (int nt = 0; nt < 12; ++nt) bb[nt] = inb[nt * 16 + m];

#pragma unroll
  for (int t = 0; t < 2; ++t) {
    int srcR[8], dstR[8];
#pragma unroll
    for (int r = 0; r < 8; ++r) {
      int eR  = e0 + t * 16 + hi * 8 + r;
      srcR[r] = edges[2 * eR];
      dstR[r] = edges[2 * eR + 1];
    }
    // phi mirror is swizzled: phi_h[n*192 + m*12 + nt]  (col = nt*16 + m)
#pragma unroll
    for (int r = 0; r < 8; ++r) {
      const _Float16* pr = phi_h + (size_t)srcR[r] * 192 + m * 12;
      v4h p0 = *(const v4h*)(pr);
      v4h p1 = *(const v4h*)(pr + 4);
      v4h p2 = *(const v4h*)(pr + 8);
#pragma unroll
      for (int nt = 0; nt < 4; ++nt) {
        acc3[t][nt][r]     = (acc3[t][nt][r]     + bb[nt])     * (float)p0[nt];
        acc3[t][nt + 4][r] = (acc3[t][nt + 4][r] + bb[nt + 4]) * (float)p1[nt];
        acc3[t][nt + 8][r] = (acc3[t][nt + 8][r] + bb[nt + 8]) * (float)p2[nt];
      }
    }
#pragma unroll
    for (int r = 0; r < 8; ++r) {
      float gp = 0.0f;
#pragma unroll
      for (int nt = 0; nt < 4; ++nt) gp += acc3[t][nt][r] * eiv[nt];
      gp += __shfl_xor(gp, 1, 32);
      gp += __shfl_xor(gp, 2, 32);
      gp += __shfl_xor(gp, 4, 32);
      gp += __shfl_xor(gp, 8, 32);          // sum within each 16-lane half
      const float gate = sigm_f(gp + eib0);

      const int eR = e0 + t * 16 + hi * 8 + r;
      const float u0 = unit[eR * 3 + 0];
      const float u1 = unit[eR * 3 + 1];
      const float u2 = unit[eR * 3 + 2];
      const int dEr = dstR[r], sEr = srcR[r];
      // v mirror is swizzled: v_h[n*192 + dim*64 + m*4 + nt]
      const _Float16* vr = v_h + (size_t)sEr * 192 + m * 4;
      v4h vd0 = *(const v4h*)(vr);
      v4h vd1 = *(const v4h*)(vr + 64);
      v4h vd2 = *(const v4h*)(vr + 128);
#pragma unroll
      for (int nt = 0; nt < 4; ++nt) {
        int col = nt * 16 + m;
        atomicAdd(&s_out[(size_t)dEr * DD + col], acc3[t][nt][r] * gate);
        float pvv = acc3[t][nt + 4][r];
        float pvs = acc3[t][nt + 8][r];
        atomicAdd(&v_out[(size_t)dEr * 192 + col],
                  ((float)vd0[nt] * pvv + pvs * u0) * gate);
        atomicAdd(&v_out[(size_t)dEr * 192 + 64 + col],
                  ((float)vd1[nt] * pvv + pvs * u1) * gate);
        atomicAdd(&v_out[(size_t)dEr * 192 + 128 + col],
                  ((float)vd2[nt] * pvv + pvs * u2) * gate);
      }
    }
  }
}

// ---------------------------------------------------------------------------
// phi = silu(sn @ msg_W1 + b1) @ msg_W2 + b2  ->  swizzled f16 mirror.
// One wave = one 16-node tile; NN % 64 == 0 so EXEC all-ones.
// ---------------------------------------------------------------------------
__global__ __launch_bounds__(128) void phi_wmma_kernel(
    const float* __restrict__ sn,
    const _Float16* __restrict__ pM1, const float* __restrict__ b1,
    const _Float16* __restrict__ pM2, const float* __restrict__ b2,
    _Float16* __restrict__ phi_h)
{
  __shared__ _Float16 hb[4][16][64];
  const int w = threadIdx.x >> 5, lane = threadIdx.x & 31;
  const int m = lane & 15, hi = lane >> 4;
  const int n0 = (blockIdx.x * 4 + w) * 16;
  const v8f zf = {0.f, 0.f, 0.f, 0.f, 0.f, 0.f, 0.f, 0.f};

  v16h a[2];
#pragma unroll
  for (int kc = 0; kc < 2; ++kc)
    a[kc] = a_from_f32(sn + (size_t)(n0 + m) * 64, kc, hi);
  v8f acc[4];
#pragma unroll
  for (int nt = 0; nt < 4; ++nt) acc[nt] = zf;
  {
    v16h bcur = loadB(pM1, 0, lane);
#pragma unroll
    for (int kc = 0; kc < 2; ++kc)
#pragma unroll
      for (int nt = 0; nt < 4; ++nt) {
        int f = kc * 4 + nt;
        v16h bnxt = loadB(pM1, (f + 1 < 8) ? f + 1 : f, lane);
        acc[nt] = WMMA(a[kc], bcur, acc[nt]);
        bcur = bnxt;
      }
  }
#pragma unroll
  for (int nt = 0; nt < 4; ++nt) {
    float b0 = b1[nt * 16 + m];
#pragma unroll
    for (int r = 0; r < 8; ++r)
      hb[w][hi * 8 + r][nt * 16 + m] = (_Float16)silu_f(acc[nt][r] + b0);
  }
  __syncthreads();

  v16h a2[2];
#pragma unroll
  for (int kc = 0; kc < 2; ++kc)
    a2[kc] = a_from_lds(&hb[w][m][0], kc, hi);
  v8f acc2[12];
#pragma unroll
  for (int nt = 0; nt < 12; ++nt) acc2[nt] = zf;
  {
    v16h bcur = loadB(pM2, 0, lane);
#pragma unroll
    for (int kc = 0; kc < 2; ++kc)
#pragma unroll
      for (int nt = 0; nt < 12; ++nt) {
        int f = kc * 12 + nt;
        v16h bnxt = loadB(pM2, (f + 1 < 24) ? f + 1 : f, lane);
        acc2[nt] = WMMA(a2[kc], bcur, acc2[nt]);
        bcur = bnxt;
      }
  }
  float bb[12];
#pragma unroll
  for (int nt = 0; nt < 12; ++nt) bb[nt] = b2[nt * 16 + m];
#pragma unroll
  for (int r = 0; r < 8; ++r) {
    _Float16* pd = phi_h + (size_t)(n0 + hi * 8 + r) * 192 + m * 12;
    v4h q0, q1, q2;
#pragma unroll
    for (int nt = 0; nt < 4; ++nt) {
      q0[nt] = (_Float16)(acc2[nt][r]     + bb[nt]);
      q1[nt] = (_Float16)(acc2[nt + 4][r] + bb[nt + 4]);
      q2[nt] = (_Float16)(acc2[nt + 8][r] + bb[nt + 8]);
    }
    *(v4h*)(pd)     = q0;
    *(v4h*)(pd + 4) = q1;
    *(v4h*)(pd + 8) = q2;
  }
}

// ---------------------------------------------------------------------------
// UV-update: UVv = v @ up_UV ; Vn = ||Vv|| ; inner = <Uv,Vv> ;
// a = silu([Vn, s] @ uv_W1 + b1) @ uv_W2 + b2 ; s,v updated in place.
// One wave = one 16-node tile; all-WMMA, no scratch spills.
// ---------------------------------------------------------------------------
__global__ __launch_bounds__(128) void update_wmma_kernel(
    const _Float16* __restrict__ pUp,
    const _Float16* __restrict__ pU1, const float* __restrict__ uvb1,
    const _Float16* __restrict__ pU2, const float* __restrict__ uvb2,
    float* __restrict__ s, _Float16* __restrict__ s_h, float* __restrict__ v)
{
  __shared__ _Float16 vnb[4][16][64];
  __shared__ _Float16 hb2[4][16][64];
  const int w = threadIdx.x >> 5, lane = threadIdx.x & 31;
  const int m = lane & 15, hi = lane >> 4;
  const int n0 = (blockIdx.x * 4 + w) * 16;
  const v8f zf = {0.f, 0.f, 0.f, 0.f, 0.f, 0.f, 0.f, 0.f};

  v8f uvacc[3][4];     // Uv fragments per spatial dim (cols 0..63)
  v8f vn2[4], inner[4];
#pragma unroll
  for (int nt = 0; nt < 4; ++nt) { vn2[nt] = zf; inner[nt] = zf; }

#pragma unroll
  for (int dim = 0; dim < 3; ++dim) {
    v16h a[2];
#pragma unroll
    for (int kc = 0; kc < 2; ++kc)
      a[kc] = a_from_f32(v + (size_t)(n0 + m) * 192 + dim * 64, kc, hi);
    v8f acc[8];
#pragma unroll
    for (int nt = 0; nt < 8; ++nt) acc[nt] = zf;
    v16h bcur = loadB(pUp, 0, lane);
#pragma unroll
    for (int kc = 0; kc < 2; ++kc)
#pragma unroll
      for (int nt = 0; nt < 8; ++nt) {
        int f = kc * 8 + nt;
        v16h bnxt = loadB(pUp, (f + 1 < 16) ? f + 1 : f, lane);
        acc[nt] = WMMA(a[kc], bcur, acc[nt]);
        bcur = bnxt;
      }
#pragma unroll
    for (int nt = 0; nt < 4; ++nt) {
      uvacc[dim][nt] = acc[nt];                        // Uv
#pragma unroll
      for (int r = 0; r < 8; ++r) {
        float vv = acc[nt + 4][r];                     // Vv
        vn2[nt][r]   += vv * vv;
        inner[nt][r] += acc[nt][r] * vv;
      }
    }
  }
#pragma unroll
  for (int nt = 0; nt < 4; ++nt)
#pragma unroll
    for (int r = 0; r < 8; ++r)
      vnb[w][hi * 8 + r][nt * 16 + m] = (_Float16)sqrtf(vn2[nt][r] + 1e-6f);
  __syncthreads();

  // a1 = silu(cat[Vn, s](16x128) @ uv_W1(128x64) + b1)
  v16h a1[4];
#pragma unroll
  for (int kc = 0; kc < 2; ++kc) {
    a1[kc]     = a_from_lds(&vnb[w][m][0], kc, hi);
    a1[kc + 2] = a_from_f32(s + (size_t)(n0 + m) * 64, kc, hi);
  }
  v8f acc1[4];
#pragma unroll
  for (int nt = 0; nt < 4; ++nt) acc1[nt] = zf;
  {
    v16h bcur = loadB(pU1, 0, lane);
#pragma unroll
    for (int kc = 0; kc < 4; ++kc)
#pragma unroll
      for (int nt = 0; nt < 4; ++nt) {
        int f = kc * 4 + nt;
        v16h bnxt = loadB(pU1, (f + 1 < 16) ? f + 1 : f, lane);
        acc1[nt] = WMMA(a1[kc], bcur, acc1[nt]);
        bcur = bnxt;
      }
  }
#pragma unroll
  for (int nt = 0; nt < 4; ++nt) {
    float b0 = uvb1[nt * 16 + m];
#pragma unroll
    for (int r = 0; r < 8; ++r)
      hb2[w][hi * 8 + r][nt * 16 + m] = (_Float16)silu_f(acc1[nt][r] + b0);
  }
  __syncthreads();

  // a2 = h(16x64) @ uv_W2(64x192)
  v16h a3[2];
#pragma unroll
  for (int kc = 0; kc < 2; ++kc)
    a3[kc] = a_from_lds(&hb2[w][m][0], kc, hi);
  v8f acc2[12];
#pragma unroll
  for (int nt = 0; nt < 12; ++nt) acc2[nt] = zf;
  {
    v16h bcur = loadB(pU2, 0, lane);
#pragma unroll
    for (int kc = 0; kc < 2; ++kc)
#pragma unroll
      for (int nt = 0; nt < 12; ++nt) {
        int f = kc * 12 + nt;
        v16h bnxt = loadB(pU2, (f + 1 < 24) ? f + 1 : f, lane);
        acc2[nt] = WMMA(a3[kc], bcur, acc2[nt]);
        bcur = bnxt;
      }
  }
  // epilogue: s += a_ss + a_sv*inner ; v += a_vv*Uv  (each element owned)
#pragma unroll
  for (int nt = 0; nt < 4; ++nt) {
    int col = nt * 16 + m;
    float bvv = uvb2[col], bsv = uvb2[64 + col], bss = uvb2[128 + col];
#pragma unroll
    for (int r = 0; r < 8; ++r) {
      int node  = n0 + hi * 8 + r;
      float avv = acc2[nt][r]     + bvv;
      float asv = acc2[nt + 4][r] + bsv;
      float ass = acc2[nt + 8][r] + bss;
      float sv  = s[(size_t)node * 64 + col] + ass + asv * inner[nt][r];
      s[(size_t)node * 64 + col]   = sv;
      s_h[(size_t)node * 64 + col] = (_Float16)sv;   // mirror for next layer
#pragma unroll
      for (int dim = 0; dim < 3; ++dim) {
        size_t vi = (size_t)node * 192 + dim * 64 + col;
        v[vi] += avv * uvacc[dim][nt][r];
      }
    }
  }
}

// ---------------------------------------------------------------------------
// Small scalar kernels
// ---------------------------------------------------------------------------
__global__ void geom_kernel(const float* __restrict__ pos, const int* __restrict__ edges,
                            float* __restrict__ dist, float* __restrict__ unit) {
  int e = blockIdx.x * blockDim.x + threadIdx.x;
  if (e >= NE) return;
  int sE = edges[2 * e], dE = edges[2 * e + 1];
  float dx = pos[dE * 3 + 0] - pos[sE * 3 + 0];
  float dy = pos[dE * 3 + 1] - pos[sE * 3 + 1];
  float dz = pos[dE * 3 + 2] - pos[sE * 3 + 2];
  float dd = sqrtf(dx * dx + dy * dy + dz * dz + 1e-12f);
  dist[e] = dd;
  float inv = 1.0f / dd;
  unit[e * 3 + 0] = dx * inv;
  unit[e * 3 + 1] = dy * inv;
  unit[e * 3 + 2] = dz * inv;
}

__global__ void proj_kernel(const float* __restrict__ x,
                            const float* __restrict__ W1, const float* __restrict__ b1,
                            const float* __restrict__ W2, const float* __restrict__ b2,
                            float* __restrict__ s, _Float16* __restrict__ s_h,
                            float* __restrict__ v) {
  int n = blockIdx.x * blockDim.x + threadIdx.x;
  if (n >= NN) return;
  float xr[32];
  for (int k = 0; k < 32; ++k) xr[k] = x[n * 32 + k];
  float h[32];
  for (int j = 0; j < 32; ++j) {
    float a = b1[j];
    for (int k = 0; k < 32; ++k) a += xr[k] * W1[k * 32 + j];
    h[j] = silu_f(a);
  }
  for (int o = 0; o < 64; ++o) {
    float a = b2[o];
    for (int k = 0; k < 32; ++k) a += h[k] * W2[k * 64 + o];
    s[n * 64 + o]   = a;
    s_h[n * 64 + o] = (_Float16)a;
  }
  for (int d = 0; d < 192; ++d) v[n * 192 + d] = 0.0f;
}

// stats layout: [gsum(16), gsum2(16), gcnt(16), gmean(16), ginv(16)]
__global__ void zero_stats_kernel(float* __restrict__ st) {
  int i = threadIdx.x;
  if (i < 5 * NG) st[i] = 0.0f;
}

__global__ void stats_kernel(const float* __restrict__ s, const int* __restrict__ gids,
                             float* __restrict__ st) {
  __shared__ float ls[NG], ls2[NG], lc[NG];
  int tid = threadIdx.x;
  if (tid < NG) { ls[tid] = 0.f; ls2[tid] = 0.f; lc[tid] = 0.f; }
  __syncthreads();
  int n = blockIdx.x * blockDim.x + tid;
  if (n < NN) {
    float a = 0.0f, a2 = 0.0f;
    for (int d = 0; d < 64; ++d) { float x = s[n * 64 + d]; a += x; a2 += x * x; }
    int g = gids[n];
    atomicAdd(&ls[g], a);
    atomicAdd(&ls2[g], a2);
    atomicAdd(&lc[g], 1.0f);
  }
  __syncthreads();
  if (tid < NG) {
    atomicAdd(&st[tid], ls[tid]);
    atomicAdd(&st[NG + tid], ls2[tid]);
    atomicAdd(&st[2 * NG + tid], lc[tid]);
  }
}

__global__ void finalize_stats_kernel(float* __restrict__ st) {
  int g = threadIdx.x;
  if (g >= NG) return;
  float cnt  = st[2 * NG + g] * 64.0f;
  float mean = st[g] / cnt;
  float ex2  = st[NG + g] / cnt;
  st[3 * NG + g] = mean;
  st[4 * NG + g] = rsqrtf(ex2 - mean * mean + 1e-5f);
}

__global__ void sn_kernel(const float* __restrict__ s, const float* __restrict__ v,
                          const int* __restrict__ gids, const float* __restrict__ st,
                          const float* __restrict__ lg, const float* __restrict__ lb,
                          float* __restrict__ sn, float* __restrict__ s_acc,
                          float* __restrict__ v_acc, _Float16* __restrict__ v_h) {
  int n = blockIdx.x * blockDim.x + threadIdx.x;
  if (n >= NN) return;
  int   g    = gids[n];
  float mean = st[3 * NG + g], inv = st[4 * NG + g];
  for (int d = 0; d < 64; ++d) {
    float x = (s[n * 64 + d] - mean) * inv * lg[d] + lb[d];
    sn[n * 64 + d]    = x;
    s_acc[n * 64 + d] = x;            // s_next = sn + segment_sum(ms)
  }
  for (int d = 0; d < 192; ++d) {
    float x = v[n * 192 + d];
    v_acc[n * 192 + d] = x;
    int dim = d >> 6, col = d & 63;
    // swizzled f16 mirror: v_h[n*192 + dim*64 + (col&15)*4 + (col>>4)]
    v_h[(size_t)n * 192 + dim * 64 + (col & 15) * 4 + (col >> 4)] = (_Float16)x;
  }
}

__global__ void copy_kernel(const float* __restrict__ src, float* __restrict__ dst, int n) {
  int i = blockIdx.x * blockDim.x + threadIdx.x;
  if (i < n) dst[i] = src[i];
}

// ---------------------------------------------------------------------------
extern "C" void kernel_launch(void* const* d_in, const int* in_sizes, int n_in,
                              void* d_out, int out_size, void* d_ws, size_t ws_size,
                              hipStream_t stream) {
  (void)in_sizes; (void)n_in; (void)out_size; (void)ws_size;

  const float* node_features  = (const float*)d_in[0];
  const float* node_positions = (const float*)d_in[1];
  const int*   edges          = (const int*)d_in[2];
  const int*   gids           = (const int*)d_in[3];
  const float* proj_W1 = (const float*)d_in[4];
  const float* proj_b1 = (const float*)d_in[5];
  const float* proj_W2 = (const float*)d_in[6];
  const float* proj_b2 = (const float*)d_in[7];
  const float* ef_W    = (const float*)d_in[8];
  const float* ef_b    = (const float*)d_in[9];
  const float* el_W1   = (const float*)d_in[10];
  const float* el_b1   = (const float*)d_in[11];
  const float* el_W2   = (const float*)d_in[12];
  const float* el_b2   = (const float*)d_in[13];
  const float* in_W    = (const float*)d_in[14];
  const float* in_b    = (const float*)d_in[15];
  const float* msg_W1  = (const float*)d_in[16];
  const float* msg_b1  = (const float*)d_in[17];
  const float* msg_W2  = (const float*)d_in[18];
  const float* msg_b2  = (const float*)d_in[19];
  const float* ei_W    = (const float*)d_in[20];
  const float* ei_b    = (const float*)d_in[21];
  const float* ln_g    = (const float*)d_in[22];
  const float* ln_b    = (const float*)d_in[23];
  const float* up_UV   = (const float*)d_in[24];
  const float* uv_W1   = (const float*)d_in[25];
  const float* uv_b1   = (const float*)d_in[26];
  const float* uv_W2   = (const float*)d_in[27];
  const float* uv_b2   = (const float*)d_in[28];

  float* ws = (float*)d_ws;
  size_t off = 0;
  float* sA    = ws + off; off += (size_t)NN * 64;
  float* sB    = ws + off; off += (size_t)NN * 64;
  float* vA    = ws + off; off += (size_t)NN * 192;
  float* vB    = ws + off; off += (size_t)NN * 192;
  float* snb   = ws + off; off += (size_t)NN * 64;
  float* distb = ws + off; off += (size_t)NE;
  float* unitb = ws + off; off += (size_t)NE * 3;
  float* stats = ws + off; off += 5 * NG;
  _Float16* packW = (_Float16*)(ws + off);       // 3 x 90112 f16
  _Float16* s_h   = packW + (size_t)3 * 90112;   // NN*64
  _Float16* v_h   = s_h + (size_t)NN * 64;       // NN*192
  _Float16* phi_h = v_h + (size_t)NN * 192;      // NN*192

  // Per-layer packed weight layout (offsets within 90112-f16 stride):
  //   pW1 24576 | pW2 8192 | pW3 12288 | pM1 4096 | pM2 12288
  //   pUp 8192  | pU1 8192 | pU2 12288
  for (int l = 0; l < NL; ++l) {
    _Float16* base = packW + (size_t)l * 90112;
    _Float16* pW1 = base;
    _Float16* pW2 = pW1 + 24576;
    _Float16* pW3 = pW2 + 8192;
    _Float16* pM1 = pW3 + 12288;
    _Float16* pM2 = pM1 + 4096;
    _Float16* pUp = pM2 + 12288;
    _Float16* pU1 = pUp + 8192;
    _Float16* pU2 = pU1 + 8192;
    pack_w_kernel<<<(24576 + 255) / 256, 256, 0, stream>>>(
        el_W1 + (size_t)l * 24576, pW1, 8, 128, 24576);
    pack_w_kernel<<<(8192 + 255) / 256, 256, 0, stream>>>(
        el_W2 + (size_t)l * 8192, pW2, 4, 64, 8192);
    pack_w_kernel<<<(12288 + 255) / 256, 256, 0, stream>>>(
        in_W + (size_t)l * 12288, pW3, 12, 192, 12288);
    pack_w_kernel<<<(4096 + 255) / 256, 256, 0, stream>>>(
        msg_W1 + (size_t)l * 4096, pM1, 4, 64, 4096);
    pack_w_kernel<<<(12288 + 255) / 256, 256, 0, stream>>>(
        msg_W2 + (size_t)l * 12288, pM2, 12, 192, 12288);
    pack_w_kernel<<<(8192 + 255) / 256, 256, 0, stream>>>(
        up_UV + (size_t)l * 8192, pUp, 8, 128, 8192);
    pack_w_kernel<<<(8192 + 255) / 256, 256, 0, stream>>>(
        uv_W1 + (size_t)l * 8192, pU1, 4, 64, 8192);
    pack_w_kernel<<<(12288 + 255) / 256, 256, 0, stream>>>(
        uv_W2 + (size_t)l * 12288, pU2, 12, 192, 12288);
  }

  geom_kernel<<<(NE + 255) / 256, 256, 0, stream>>>(node_positions, edges, distb, unitb);
  proj_kernel<<<(NN + 255) / 256, 256, 0, stream>>>(node_features, proj_W1, proj_b1,
                                                    proj_W2, proj_b2, sA, s_h, vA);

  float* s_cur = sA; float* v_cur = vA;
  float* s_nxt = sB; float* v_nxt = vB;

  for (int l = 0; l < NL; ++l) {
    _Float16* base = packW + (size_t)l * 90112;
    _Float16* pW1 = base;
    _Float16* pW2 = pW1 + 24576;
    _Float16* pW3 = pW2 + 8192;
    _Float16* pM1 = pW3 + 12288;
    _Float16* pM2 = pM1 + 4096;
    _Float16* pUp = pM2 + 12288;
    _Float16* pU1 = pUp + 8192;
    _Float16* pU2 = pU1 + 8192;

    zero_stats_kernel<<<1, 128, 0, stream>>>(stats);
    stats_kernel<<<(NN + 255) / 256, 256, 0, stream>>>(s_cur, gids, stats);
    finalize_stats_kernel<<<1, NG, 0, stream>>>(stats);
    sn_kernel<<<(NN + 255) / 256, 256, 0, stream>>>(s_cur, v_cur, gids, stats,
                                                    ln_g + l * 64, ln_b + l * 64,
                                                    snb, s_nxt, v_nxt, v_h);
    phi_wmma_kernel<<<NN / 64, 128, 0, stream>>>(snb, pM1, msg_b1 + l * 64,
                                                 pM2, msg_b2 + l * 192, phi_h);
    edge_kernel<<<NE / 128, 128, 0, stream>>>(s_h, v_h, phi_h, distb, unitb, edges,
        ef_W + l * 64, ef_b + l * 64,
        pW1, el_b1 + l * 128,
        pW2, el_b2 + l * 64,
        pW3, in_b + l * 192,
        ei_W + l * 64, ei_b + l,
        s_nxt, v_nxt);
    update_wmma_kernel<<<NN / 64, 128, 0, stream>>>(pUp,
                                                    pU1, uv_b1 + l * 64,
                                                    pU2, uv_b2 + l * 192,
                                                    s_nxt, s_h, v_nxt);
    float* ts = s_cur; s_cur = s_nxt; s_nxt = ts;
    float* tv = v_cur; v_cur = v_nxt; v_nxt = tv;
  }

  copy_kernel<<<(NN * 64 + 255) / 256, 256, 0, stream>>>(s_cur, (float*)d_out, NN * 64);
}
